// MyQwen3SparseMLP_16569983828102
// MI455X (gfx1250) — matrix operand order repaired and checked by reading, not compiled, not run on hardware
//
#include <hip/hip_runtime.h>
#include <hip/hip_bf16.h>

// ---------------------------------------------------------------------------
// MoE sparse MLP (T=4096, D=768, I=2048, E=8), one expert per token.
//   y_t = (silu(x_t G_e^T) * (x_t U_e^T)) D_e^T,  e = position_index[t]
//
// MI455X roofline: ~38.7 GFLOP, ~176MB fp32 working set (fits 192MB L2) ->
// compute/L2-bound; use v_wmma_f32_16x16x32_bf16 (bf16 in, fp32 accumulate).
//   Phase 0 (if scratch allows): one-shot fp32->bf16 convert of x + weights
//            (halves L2 GEMM traffic, removes all per-fragment cvt VALU work).
//   Phase 1: route tokens into per-expert buckets + tile table.
//   Phase 2: fused gate+up GEMM -> H = silu(g)*u (bf16, scratch, L2-resident).
//   Phase 3: down GEMM, scatter rows to out[token].
// WMMA fragments are built directly from global memory using a shared
// contiguous K order (k = 16*kh + s) for A and B; the torch [out,in] weight layout means B fragments
// are K-contiguous loads (no transpose instructions needed).
// ---------------------------------------------------------------------------

#define T_TOK 4096
#define D_DIM 768
#define I_DIM 2048
#define E_NUM 8
#define MAX_TILES 48   // sum_e ceil(Te/128) <= 32 + 7 = 39

// meta layout (ints) at d_ws + 0:
//   [0] numTiles; [1..9] offsets[0..8]; [16..79] tileExpert;
//   [80..143] tilePosBase; [144..4239] bucket[4096]
#define META_BYTES   17408u                       // 256B-aligned region
#define XB_OFF       (META_BYTES)                                  // bf16 x
#define GB_OFF       (XB_OFF + (size_t)T_TOK * D_DIM * 2)          // bf16 gate
#define UB_OFF       (GB_OFF + (size_t)E_NUM * I_DIM * D_DIM * 2)  // bf16 up
#define DB_OFF       (UB_OFF + (size_t)E_NUM * I_DIM * D_DIM * 2)  // bf16 down
#define HF_OFF       (DB_OFF + (size_t)E_NUM * I_DIM * D_DIM * 2)  // H (full)
#define WS_FULL      (HF_OFF + (size_t)T_TOK * I_DIM * 2)          // ~98.6 MB
#define HM_OFF       (META_BYTES)                                  // H (min path)
#define WS_MIN       (HM_OFF + (size_t)T_TOK * I_DIM * 2)          // ~16.8 MB

typedef __bf16 bf16_t;
typedef __attribute__((ext_vector_type(16))) __bf16 v16bf;
typedef __attribute__((ext_vector_type(8)))  __bf16 v8bf;
typedef __attribute__((ext_vector_type(8)))  float  v8f;
typedef __attribute__((ext_vector_type(8)))  float  f32x8;

// ---- fragment builders (contiguous K order, same for A and B) ----------------
// A (16x32 bf16): lane L holds row M=L&15; K-half (L>>4) selects one
//   contiguous 16-element K run at [k0+kh*16).
// B (32x16 bf16): lane L holds col N=L&15; K-half selects one contiguous
//   16-element K run at [k0+kh*16).

template <typename T>
__device__ __forceinline__ v16bf fragK2x8(const T* p0, const T* p1);

template <>
__device__ __forceinline__ v16bf fragK2x8<float>(const float* p0, const float* p1) {
  v8bf lo = __builtin_convertvector(*(const f32x8*)p0, v8bf);
  v8bf hi = __builtin_convertvector(*(const f32x8*)p1, v8bf);
  return __builtin_shufflevector(lo, hi, 0,1,2,3,4,5,6,7,8,9,10,11,12,13,14,15);
}
template <>
__device__ __forceinline__ v16bf fragK2x8<bf16_t>(const bf16_t* p0, const bf16_t* p1) {
  v8bf lo = *(const v8bf*)p0;
  v8bf hi = *(const v8bf*)p1;
  return __builtin_shufflevector(lo, hi, 0,1,2,3,4,5,6,7,8,9,10,11,12,13,14,15);
}

template <typename T>
__device__ __forceinline__ v16bf fragK16(const T* p) { return fragK2x8<T>(p, p + 8); }

__device__ __forceinline__ v8f wmma_bf16(v16bf a, v16bf b, v8f c) {
  // (neg_a, A, neg_b, B, c_mod, C, reuse_a, reuse_b)
  return __builtin_amdgcn_wmma_f32_16x16x32_bf16(false, a, false, b, (short)0, c,
                                                 false, false);
}

// ---------------------------------------------------------------------------
// Phase 0: bulk fp32 -> bf16 convert (bandwidth-bound, ~10us at 23.3 TB/s).
__global__ __launch_bounds__(256) void cvt_kernel(const float* __restrict__ src,
                                                  bf16_t* __restrict__ dst, int n8) {
  for (int i = blockIdx.x * blockDim.x + threadIdx.x; i < n8;
       i += gridDim.x * blockDim.x) {
    const f32x8 v = *(const f32x8*)(src + (size_t)i * 8);
    *(v8bf*)(dst + (size_t)i * 8) = __builtin_convertvector(v, v8bf);
  }
}

// ---------------------------------------------------------------------------
__global__ __launch_bounds__(256) void route_kernel(const int* __restrict__ pos,
                                                    int* __restrict__ meta) {
  __shared__ int cnt[E_NUM];
  __shared__ int off[E_NUM + 1];
  __shared__ int cur[E_NUM];
  const int t = threadIdx.x;
  if (t < E_NUM) cnt[t] = 0;
  __syncthreads();
  for (int i = t; i < T_TOK; i += 256) {
    atomicAdd(&cnt[pos[i] & (E_NUM - 1)], 1);
  }
  __syncthreads();
  if (t == 0) {
    int s = 0, nt = 0;
    for (int e = 0; e < E_NUM; ++e) { off[e] = s; s += cnt[e]; }
    off[E_NUM] = s;
    for (int e = 0; e < E_NUM; ++e) {
      for (int mb = 0; mb < cnt[e]; mb += 128) {
        meta[16 + nt] = e;
        meta[80 + nt] = off[e] + mb;
        ++nt;
      }
    }
    meta[0] = nt;
    for (int e = 0; e <= E_NUM; ++e) meta[1 + e] = off[e];
  }
  __syncthreads();
  if (t < E_NUM) cur[t] = off[t];
  __syncthreads();
  for (int i = t; i < T_TOK; i += 256) {
    const int e = pos[i] & (E_NUM - 1);
    const int r = atomicAdd(&cur[e], 1);
    meta[144 + r] = i;   // bucket[r] = token index
  }
}

// ---------------------------------------------------------------------------
// Phase 2: H[p, :] = silu(x G^T) * (x U^T) for bucketed token rows.
// Block = 256 thr (8 waves). Tile: 128 tokens x 64 I-cols. Wave slab: 32x32.
template <typename TW>
__global__ __launch_bounds__(256) void gateup_kernel(
    const TW* __restrict__ x, const TW* __restrict__ gw,
    const TW* __restrict__ uw, const int* __restrict__ meta,
    bf16_t* __restrict__ H) {
  const int tile = blockIdx.y;
  if (tile >= meta[0]) return;
  const int e       = meta[16 + tile];
  const int posBase = meta[80 + tile];
  const int posEnd  = meta[2 + e];       // offsets[e+1]
  const int* bucket = meta + 144;

  const int lane  = threadIdx.x & 31;
  const int w     = threadIdx.x >> 5;
  const int waveM = w & 3;               // 4 waves along tokens (32 rows each)
  const int waveN = w >> 2;              // 2 waves along I cols (32 cols each)
  const int mBase = posBase + waveM * 32;
  const int nBase = blockIdx.x * 64 + waveN * 32;

  const TW* GW = gw + (size_t)e * I_DIM * D_DIM;
  const TW* UW = uw + (size_t)e * I_DIM * D_DIM;

  const int kh   = lane >> 4;
  const int mrow = lane & 15;
  // Gather token rows for this lane's two A fragments (bucket fully populated;
  // rows past 4095 only occur in the clamp tail and are masked on store).
  const int t0 = bucket[min(mBase + mrow,      T_TOK - 1)];
  const int t1 = bucket[min(mBase + 16 + mrow, T_TOK - 1)];
  const TW* xr0 = x + (size_t)t0 * D_DIM;
  const TW* xr1 = x + (size_t)t1 * D_DIM;
  // B fragment rows: weight row i = output col n (row-major [I, D] layout).
  const TW* g0r = GW + (size_t)(nBase + mrow)      * D_DIM;
  const TW* g1r = GW + (size_t)(nBase + 16 + mrow) * D_DIM;
  const TW* u0r = UW + (size_t)(nBase + mrow)      * D_DIM;
  const TW* u1r = UW + (size_t)(nBase + 16 + mrow) * D_DIM;

  v8f aG[2][2] = {};
  v8f aU[2][2] = {};

  for (int k0 = 0; k0 < D_DIM; k0 += 32) {
    __builtin_prefetch(g0r + k0 + 32, 0, 1);   // speculative: dropped if OOB
    __builtin_prefetch(u0r + k0 + 32, 0, 1);
    const v16bf a0  = fragK16<TW>(xr0 + k0 + kh * 16);
    const v16bf a1  = fragK16<TW>(xr1 + k0 + kh * 16);
    const v16bf bg0 = fragK16<TW>(g0r + k0 + kh * 16);
    const v16bf bg1 = fragK16<TW>(g1r + k0 + kh * 16);
    const v16bf bu0 = fragK16<TW>(u0r + k0 + kh * 16);
    const v16bf bu1 = fragK16<TW>(u1r + k0 + kh * 16);
    aG[0][0] = wmma_bf16(a0, bg0, aG[0][0]);
    aG[0][1] = wmma_bf16(a0, bg1, aG[0][1]);
    aG[1][0] = wmma_bf16(a1, bg0, aG[1][0]);
    aG[1][1] = wmma_bf16(a1, bg1, aG[1][1]);
    aU[0][0] = wmma_bf16(a0, bu0, aU[0][0]);
    aU[0][1] = wmma_bf16(a0, bu1, aU[0][1]);
    aU[1][0] = wmma_bf16(a1, bu0, aU[1][0]);
    aU[1][1] = wmma_bf16(a1, bu1, aU[1][1]);
  }

  // C/D layout: lanes 0-15 hold M=j (VGPR j), lanes 16-31 hold M=j+8; N=lane&15.
  const int half = lane >> 4;
  const int ncol = lane & 15;
  #pragma unroll
  for (int mf = 0; mf < 2; ++mf) {
    #pragma unroll
    for (int nf = 0; nf < 2; ++nf) {
      const v8f g = aG[mf][nf];
      const v8f u = aU[mf][nf];
      #pragma unroll
      for (int j = 0; j < 8; ++j) {
        const int p = mBase + mf * 16 + j + 8 * half;
        if (p < posEnd) {
          const float gg = g[j];
          const float h  = (gg / (1.0f + __expf(-gg))) * u[j];
          H[(size_t)p * I_DIM + nBase + nf * 16 + ncol] = (__bf16)h;
        }
      }
    }
  }
}

// ---------------------------------------------------------------------------
// Phase 3: out[token[p], :] = H[p, :] @ down^T.  Tile 128 x 64, wave slab 32x32.
template <typename TW>
__global__ __launch_bounds__(256) void down_kernel(
    const bf16_t* __restrict__ H, const TW* __restrict__ dw,
    const int* __restrict__ meta, float* __restrict__ out) {
  const int tile = blockIdx.y;
  if (tile >= meta[0]) return;
  const int e       = meta[16 + tile];
  const int posBase = meta[80 + tile];
  const int posEnd  = meta[2 + e];
  const int* bucket = meta + 144;

  const int lane  = threadIdx.x & 31;
  const int w     = threadIdx.x >> 5;
  const int waveM = w & 3;
  const int waveN = w >> 2;
  const int mBase = posBase + waveM * 32;
  const int nBase = blockIdx.x * 64 + waveN * 32;   // output D col

  const TW* DW = dw + (size_t)e * D_DIM * I_DIM;

  const int kh   = lane >> 4;
  const int mrow = lane & 15;
  const bf16_t* h0 = H + (size_t)min(mBase + mrow,      T_TOK - 1) * I_DIM;
  const bf16_t* h1 = H + (size_t)min(mBase + 16 + mrow, T_TOK - 1) * I_DIM;
  const TW* d0r = DW + (size_t)(nBase + mrow)      * I_DIM;
  const TW* d1r = DW + (size_t)(nBase + 16 + mrow) * I_DIM;

  v8f acc[2][2] = {};
  for (int k0 = 0; k0 < I_DIM; k0 += 32) {
    __builtin_prefetch(d0r + k0 + 32, 0, 1);
    const v16bf a0 = fragK16<bf16_t>(h0 + k0 + kh * 16);
    const v16bf a1 = fragK16<bf16_t>(h1 + k0 + kh * 16);
    const v16bf b0 = fragK16<TW>(d0r + k0 + kh * 16);
    const v16bf b1 = fragK16<TW>(d1r + k0 + kh * 16);
    acc[0][0] = wmma_bf16(a0, b0, acc[0][0]);
    acc[0][1] = wmma_bf16(a0, b1, acc[0][1]);
    acc[1][0] = wmma_bf16(a1, b0, acc[1][0]);
    acc[1][1] = wmma_bf16(a1, b1, acc[1][1]);
  }

  const int half = lane >> 4;
  const int ncol = lane & 15;
  #pragma unroll
  for (int mf = 0; mf < 2; ++mf) {
    #pragma unroll
    for (int nf = 0; nf < 2; ++nf) {
      const v8f c = acc[mf][nf];
      #pragma unroll
      for (int j = 0; j < 8; ++j) {
        const int p = mBase + mf * 16 + j + 8 * half;
        if (p < posEnd) {
          const int tok = bucket[p];               // each token in exactly one bucket
          out[(size_t)tok * D_DIM + nBase + nf * 16 + ncol] = c[j];
        }
      }
    }
  }
}

// ---------------------------------------------------------------------------
extern "C" void kernel_launch(void* const* d_in, const int* in_sizes, int n_in,
                              void* d_out, int out_size, void* d_ws, size_t ws_size,
                              hipStream_t stream) {
  (void)in_sizes; (void)n_in; (void)out_size;
  const float* x   = (const float*)d_in[0];
  const int*   pos = (const int*)d_in[1];
  // d_in[2] = behavior_index (unused: behavior_injection=False)
  const float* gw  = (const float*)d_in[3];
  const float* uw  = (const float*)d_in[4];
  const float* dw  = (const float*)d_in[5];

  int* meta = (int*)d_ws;
  route_kernel<<<dim3(1), dim3(256), 0, stream>>>(pos, meta);

  const dim3 g2(I_DIM / 64, MAX_TILES);
  const dim3 g3(D_DIM / 64, MAX_TILES);
  const dim3 blk(256);

  if (ws_size >= WS_FULL) {
    // Pre-convert operands to bf16 once; GEMM loops then run cvt-free with
    // half the L2 traffic (weights dominate and stay L2-resident).
    bf16_t* xb = (bf16_t*)((char*)d_ws + XB_OFF);
    bf16_t* gb = (bf16_t*)((char*)d_ws + GB_OFF);
    bf16_t* ub = (bf16_t*)((char*)d_ws + UB_OFF);
    bf16_t* db = (bf16_t*)((char*)d_ws + DB_OFF);
    bf16_t* H  = (bf16_t*)((char*)d_ws + HF_OFF);
    const int nw8 = (E_NUM * I_DIM * D_DIM) / 8;   // 1,572,864
    const int nx8 = (T_TOK * D_DIM) / 8;           //   393,216
    cvt_kernel<<<dim3(1024), blk, 0, stream>>>(x,  xb, nx8);
    cvt_kernel<<<dim3(2048), blk, 0, stream>>>(gw, gb, nw8);
    cvt_kernel<<<dim3(2048), blk, 0, stream>>>(uw, ub, nw8);
    cvt_kernel<<<dim3(2048), blk, 0, stream>>>(dw, db, nw8);
    gateup_kernel<bf16_t><<<g2, blk, 0, stream>>>(xb, gb, ub, meta, H);
    down_kernel<bf16_t><<<g3, blk, 0, stream>>>(H, db, meta, (float*)d_out);
  } else {
    // Minimal-scratch fallback: convert fragments on load.
    bf16_t* H = (bf16_t*)((char*)d_ws + HM_OFF);
    gateup_kernel<float><<<g2, blk, 0, stream>>>(x, gw, uw, meta, H);
    down_kernel<float><<<g3, blk, 0, stream>>>(H, dw, meta, (float*)d_out);
  }
}
